// MambaPolarDecoder_30966714204765
// MI455X (gfx1250) — matrix operand order repaired and checked by reading, not compiled, not run on hardware
//
#include <hip/hip_runtime.h>
#include <hip/hip_bf16.h>
#include <math.h>

typedef __attribute__((ext_vector_type(16))) _Float16 v16h;
typedef __attribute__((ext_vector_type(8)))  _Float16 v8h;
typedef __attribute__((ext_vector_type(8)))  float    v8f;

#define BATCH   1024
#define SEQ     32
#define DMODEL  64
#define NLAYER  4
#define DSTATE  16
#define DCONV   4
#define DINNER  128
#define DTRANK  4

// ---------------- workspace layout (in halves) ----------------
#define WS_INW_OFF     0
#define WS_INW_N       (DMODEL * 3 * DMODEL)            // 64*192
#define WS_INPROJ_OFF  (WS_INW_OFF + WS_INW_N)
#define WS_INPROJ_N    (NLAYER * 2 * 2 * DINNER * DMODEL) // 8*256*64
#define WS_XPROJ_OFF   (WS_INPROJ_OFF + WS_INPROJ_N)
#define WS_XPROJ_N     (NLAYER * 2 * 48 * DINNER)        // 8*48*128 (padded 36->48)
#define WS_OUTPROJ_OFF (WS_XPROJ_OFF + WS_XPROJ_N)
#define WS_OUTPROJ_N   (NLAYER * 2 * DMODEL * DINNER)    // 8*64*128
#define WS_TOTAL       (WS_OUTPROJ_OFF + WS_OUTPROJ_N)

// Convert weights to f16 (and pad xproj N 36->48) once per launch.
__global__ void prep_weights(const float* __restrict__ in_w,
                             const float* __restrict__ m_in_proj,
                             const float* __restrict__ m_xproj,
                             const float* __restrict__ m_out_proj,
                             _Float16* __restrict__ ws) {
  for (int i = blockIdx.x * blockDim.x + threadIdx.x; i < WS_TOTAL;
       i += gridDim.x * blockDim.x) {
    float v;
    if (i < WS_INPROJ_OFF) {
      v = in_w[i - WS_INW_OFF];
    } else if (i < WS_XPROJ_OFF) {
      v = m_in_proj[i - WS_INPROJ_OFF];
    } else if (i < WS_OUTPROJ_OFF) {
      int j   = i - WS_XPROJ_OFF;
      int ld  = j / (48 * DINNER);
      int rem = j % (48 * DINNER);
      int row = rem / DINNER;
      int k   = rem % DINNER;
      v = (row < 36) ? m_xproj[(ld * 36 + row) * DINNER + k] : 0.0f;
    } else {
      v = m_out_proj[i - WS_OUTPROJ_OFF];
    }
    ws[i] = (_Float16)v;
  }
}

// butterfly sum across wave32; every lane gets the total
__device__ __forceinline__ float wave_sum(float v) {
#pragma unroll
  for (int off = 16; off > 0; off >>= 1) v += __shfl_xor(v, off, 32);
  return v;
}

// One 16x16 output tile of  out[m0.., n0..] += A(MxK,f16 LDS) * W(NxK,f16 global)^T
// A fragment per 16-bit A 16x32 layout, B fragment per 16-bit B 32x16 layout.
__device__ __forceinline__ v8f gemm_tile(const _Float16* __restrict__ aLds, int lda,
                                         int m0, bool flip,
                                         const _Float16* __restrict__ bGlb, int ldb,
                                         int n0, int K, int lane) {
  v8f acc = {};
  int arow = m0 + (lane & 15);
  if (flip) arow = (SEQ - 1) - arow;
  int bn = n0 + (lane & 15);
  int hi = lane >> 4;
  for (int k0 = 0; k0 < K; k0 += 32) {
    const v8h* ap0 = (const v8h*)(aLds + arow * lda + k0 + 8 * hi);
    const v8h* ap1 = (const v8h*)(aLds + arow * lda + k0 + 8 * hi + 16);
    v8h alo = ap0[0];
    v8h ahi = ap1[0];
    const v8h* bp = (const v8h*)(bGlb + bn * ldb + k0 + 8 * hi);
    v8h blo = bp[0];
    v8h bhi = bp[2];
    v16h a, bfr;
#pragma unroll
    for (int i = 0; i < 8; i++) {
      a[i] = alo[i];  a[i + 8] = ahi[i];
      bfr[i] = blo[i]; bfr[i + 8] = bhi[i];
    }
    acc = __builtin_amdgcn_wmma_f32_16x16x32_f16(false, a, false, bfr,
                                                 (short)0, acc, false, false);
  }
  return acc;
}

__device__ __forceinline__ float silu_f(float x) {
  return x / (1.0f + __expf(-x));
}

// ---------------- fused model kernel: one block per batch element ------------
__global__ __launch_bounds__(256)
void mamba_polar_fused(const float* __restrict__ channel,     // [B,32]
                       const float* __restrict__ snr_db,      // [B]
                       const int*   __restrict__ frozen,      // [B,32]
                       const float* __restrict__ emb_table,   // [2,64]
                       const float* __restrict__ lin1_w,      // [64]
                       const float* __restrict__ lin1_b,      // [64]
                       const float* __restrict__ lin2_w,      // [64]
                       const float* __restrict__ lin2_b,      // [64]
                       const float* __restrict__ in_b,        // [64]
                       const float* __restrict__ bim_ln_w,    // [L,64]
                       const float* __restrict__ bim_ln_b,    // [L,64]
                       const float* __restrict__ m_conv_w,    // [L,2,128,4]
                       const float* __restrict__ m_conv_b,    // [L,2,128]
                       const float* __restrict__ m_dt_w,      // [L,2,128,4]
                       const float* __restrict__ m_dt_b,      // [L,2,128]
                       const float* __restrict__ m_A_log,     // [L,2,128,16]
                       const float* __restrict__ m_D,         // [L,2,128]
                       const float* __restrict__ post_ln_w,   // [64]
                       const float* __restrict__ post_ln_b,   // [64]
                       const float* __restrict__ final_w,     // [64]
                       const float* __restrict__ final_b,     // [1]
                       const _Float16* __restrict__ wsInW,    // [64,192]
                       const _Float16* __restrict__ wsInProj, // [L*2,256,64]
                       const _Float16* __restrict__ wsXproj,  // [L*2,48,128]
                       const _Float16* __restrict__ wsOutProj,// [L*2,64,128]
                       float* __restrict__ out) {             // [B,32]
  // LDS carve-out: 12800 floats = 50 KB
  __shared__ __align__(16) float sm[12800];
  float*    XE  = sm;                           // [32][64] f32 residual stream
  _Float16* X0h = (_Float16*)(sm + 2048);       // [32][64] f16 initial x
  _Float16* H16 = (_Float16*)(sm + 3072);       // [32][64] f16 LN(xe)
  float*    R1  = sm + 4096;                    // 16KB: concat16 / raw xc f32 / dt f32
  _Float16* Z16 = (_Float16*)(sm + 8192);       // [32][128] f16 gate input z
  _Float16* U16 = (_Float16*)(sm + 9216);       // [32][128] f16 conv+silu output
  _Float16* Y16 = (_Float16*)(sm + 10240);      // [32][128] f16 gated scan output
  float*    DBL = sm + 11264;                   // [32][48]  f32 xproj output

  const int b    = blockIdx.x;
  const int tid  = threadIdx.x;
  const int lane = tid & 31;
  const int wave = tid >> 5;

  // ---------------- stage 0: embeddings + input projection ----------------
  {
    _Float16* C16 = (_Float16*)R1;              // [32][192] f16 concat
    const float* ch = channel + b * SEQ;
    const float  snr = snr_db[b];
    const int*   fz = frozen + b * SEQ;
    for (int i = tid; i < SEQ * 192; i += 256) {
      int t = i / 192, j = i % 192;
      float v;
      if (j < 64)       v = ch[t] * lin1_w[j] + lin1_b[j];
      else if (j < 128) v = snr * lin2_w[j - 64] + lin2_b[j - 64];
      else              v = emb_table[fz[t] * 64 + (j - 128)];
      C16[t * 192 + j] = (_Float16)v;
    }
    __syncthreads();
    // GEMM 32x64, K=192; 8 tiles -> one per wave
    int m0 = (wave & 1) * 16, n0 = (wave >> 1) * 16;
    v8f acc = gemm_tile(C16, 192, m0, false, wsInW, 192, n0, 192, lane);
    int n = n0 + (lane & 15), hi = lane >> 4;
    float bias = in_b[n];
#pragma unroll
    for (int r = 0; r < 8; r++) {
      int m = m0 + r + 8 * hi;
      float v = acc[r] + bias;
      XE[m * 64 + n]  = v;
      X0h[m * 64 + n] = (_Float16)v;
    }
  }
  __syncthreads();

  // ---------------- layers ----------------
  for (int l = 0; l < NLAYER; l++) {
    // LN(xe) -> H16 (shared by both directions)
#pragma unroll
    for (int rr = 0; rr < 4; rr++) {
      int row = wave * 4 + rr;
      float v0 = XE[row * 64 + lane];
      float v1 = XE[row * 64 + lane + 32];
      float mean = wave_sum(v0 + v1) * (1.0f / 64.0f);
      float d0 = v0 - mean, d1 = v1 - mean;
      float var = wave_sum(d0 * d0 + d1 * d1) * (1.0f / 64.0f);
      float rs = rsqrtf(var + 1e-5f);
      H16[row * 64 + lane]      = (_Float16)(d0 * rs * bim_ln_w[l * 64 + lane]      + bim_ln_b[l * 64 + lane]);
      H16[row * 64 + lane + 32] = (_Float16)(d1 * rs * bim_ln_w[l * 64 + lane + 32] + bim_ln_b[l * 64 + lane + 32]);
    }
    __syncthreads();

    for (int dir = 0; dir < 2; dir++) {
      const int ld = l * 2 + dir;

      // ---- in_proj: [32x64] @ [256x64]^T -> xc(f32 R1) | z(f16 Z16) ----
      {
        const _Float16* Wip = wsInProj + ld * 2 * DINNER * DMODEL;
#pragma unroll
        for (int i = 0; i < 4; i++) {
          int tdx = wave * 4 + i;
          int m0 = (tdx & 1) * 16, n0 = (tdx >> 1) * 16;
          v8f acc = gemm_tile(H16, 64, m0, dir == 1, Wip, 64, n0, 64, lane);
          int n = n0 + (lane & 15), hi = lane >> 4;
#pragma unroll
          for (int r = 0; r < 8; r++) {
            int m = m0 + r + 8 * hi;
            if (n < DINNER) R1[m * DINNER + n] = acc[r];
            else            Z16[m * DINNER + (n - DINNER)] = (_Float16)acc[r];
          }
        }
      }
      __syncthreads();

      // ---- depthwise causal conv (k=4) + SiLU -> U16 ----
      for (int i = tid; i < SEQ * DINNER; i += 256) {
        int t = i >> 7, d = i & 127;
        float a = m_conv_b[ld * DINNER + d];
#pragma unroll
        for (int k = 0; k < DCONV; k++) {
          int tt = t - (DCONV - 1) + k;
          if (tt >= 0) a += R1[tt * DINNER + d] * m_conv_w[(ld * DINNER + d) * DCONV + k];
        }
        U16[t * DINNER + d] = (_Float16)silu_f(a);
      }
      __syncthreads();

      // ---- xproj: [32x128] @ [48x128]^T -> DBL[32][48] (rows 36..47 zero) ----
      if (wave < 6) {
        const _Float16* Wxp = wsXproj + ld * 48 * DINNER;
        int m0 = (wave & 1) * 16, n0 = (wave >> 1) * 16;
        v8f acc = gemm_tile(U16, DINNER, m0, false, Wxp, DINNER, n0, DINNER, lane);
        int n = n0 + (lane & 15), hi = lane >> 4;
#pragma unroll
        for (int r = 0; r < 8; r++) {
          int m = m0 + r + 8 * hi;
          DBL[m * 48 + n] = acc[r];
        }
      }
      __syncthreads();

      // ---- dt = softplus(dbl[:, :4] @ dt_w^T + dt_b) -> R1 (reused) ----
      for (int i = tid; i < SEQ * DINNER; i += 256) {
        int t = i >> 7, d = i & 127;
        float a = m_dt_b[ld * DINNER + d];
#pragma unroll
        for (int r = 0; r < DTRANK; r++)
          a += DBL[t * 48 + r] * m_dt_w[(ld * DINNER + d) * DTRANK + r];
        a = (a > 20.0f) ? a : log1pf(__expf(a));
        R1[t * DINNER + d] = a;
      }
      __syncthreads();

      // ---- selective scan fused with D-skip + SiLU(z) gate -> Y16 ----
      {
        int d  = wave * 16 + (lane >> 1);   // channel owned by this lane pair
        int sh = (lane & 1) * 8;            // state half
        float Ac[8], h[8];
#pragma unroll
        for (int j = 0; j < 8; j++) {
          Ac[j] = -__expf(m_A_log[(ld * DINNER + d) * DSTATE + sh + j]);
          h[j] = 0.0f;
        }
        float Dd = m_D[ld * DINNER + d];
        for (int t = 0; t < SEQ; t++) {
          float dtv = R1[t * DINNER + d];
          float uv  = (float)U16[t * DINNER + d];
          float du  = dtv * uv;
          const float* Bp = DBL + t * 48 + DTRANK;
          const float* Cp = DBL + t * 48 + DTRANK + DSTATE;
          float acc = 0.0f;
#pragma unroll
          for (int j = 0; j < 8; j++) {
            float dA = __expf(dtv * Ac[j]);
            h[j] = dA * h[j] + du * Bp[sh + j];
            acc += h[j] * Cp[sh + j];
          }
          acc += __shfl_xor(acc, 1, 32);
          if ((lane & 1) == 0) {
            float zv = (float)Z16[t * DINNER + d];
            float yv = (acc + Dd * uv) * silu_f(zv);
            Y16[t * DINNER + d] = (_Float16)yv;
          }
        }
      }
      __syncthreads();

      // ---- out_proj: [32x128] @ [64x128]^T, accumulate into XE (flip if bwd) ----
      {
        const _Float16* Wop = wsOutProj + ld * DMODEL * DINNER;
        int m0 = (wave & 1) * 16, n0 = (wave >> 1) * 16;
        v8f acc = gemm_tile(Y16, DINNER, m0, false, Wop, DINNER, n0, DINNER, lane);
        int n = n0 + (lane & 15), hi = lane >> 4;
#pragma unroll
        for (int r = 0; r < 8; r++) {
          int m = m0 + r + 8 * hi;
          int trow = dir ? (SEQ - 1 - m) : m;
          XE[trow * 64 + n] += acc[r];
        }
      }
      __syncthreads();
    } // dir
  } // layers

  // ---------------- final: residual + post LN + linear head ----------------
#pragma unroll
  for (int rr = 0; rr < 4; rr++) {
    int row = wave * 4 + rr;
    float v0 = XE[row * 64 + lane]      + (float)X0h[row * 64 + lane];       // RESIDUAL_SCALE = 1
    float v1 = XE[row * 64 + lane + 32] + (float)X0h[row * 64 + lane + 32];
    float mean = wave_sum(v0 + v1) * (1.0f / 64.0f);
    float d0 = v0 - mean, d1 = v1 - mean;
    float var = wave_sum(d0 * d0 + d1 * d1) * (1.0f / 64.0f);
    float rs = rsqrtf(var + 1e-5f);
    float n0v = d0 * rs * post_ln_w[lane]      + post_ln_b[lane];
    float n1v = d1 * rs * post_ln_w[lane + 32] + post_ln_b[lane + 32];
    float dot = wave_sum(n0v * final_w[lane] + n1v * final_w[lane + 32]);
    if (lane == 0) out[b * SEQ + row] = dot + final_b[0];
  }
}

extern "C" void kernel_launch(void* const* d_in, const int* in_sizes, int n_in,
                              void* d_out, int out_size, void* d_ws, size_t ws_size,
                              hipStream_t stream) {
  // setup_inputs() dict order
  const float* channel   = (const float*)d_in[0];
  const float* snr_db    = (const float*)d_in[1];
  const int*   frozen    = (const int*)  d_in[2];
  const float* emb_table = (const float*)d_in[3];
  const float* lin1_w    = (const float*)d_in[4];
  const float* lin1_b    = (const float*)d_in[5];
  const float* lin2_w    = (const float*)d_in[6];
  const float* lin2_b    = (const float*)d_in[7];
  const float* in_w      = (const float*)d_in[8];
  const float* in_b      = (const float*)d_in[9];
  const float* bim_ln_w  = (const float*)d_in[10];
  const float* bim_ln_b  = (const float*)d_in[11];
  const float* m_in_proj = (const float*)d_in[12];
  const float* m_conv_w  = (const float*)d_in[13];
  const float* m_conv_b  = (const float*)d_in[14];
  const float* m_xproj   = (const float*)d_in[15];
  const float* m_dt_w    = (const float*)d_in[16];
  const float* m_dt_b    = (const float*)d_in[17];
  const float* m_A_log   = (const float*)d_in[18];
  const float* m_D       = (const float*)d_in[19];
  const float* m_out_proj= (const float*)d_in[20];
  const float* post_ln_w = (const float*)d_in[21];
  const float* post_ln_b = (const float*)d_in[22];
  const float* final_w   = (const float*)d_in[23];
  const float* final_b   = (const float*)d_in[24];

  _Float16* ws = (_Float16*)d_ws;

  prep_weights<<<256, 256, 0, stream>>>(in_w, m_in_proj, m_xproj, m_out_proj, ws);

  mamba_polar_fused<<<BATCH, 256, 0, stream>>>(
      channel, snr_db, frozen, emb_table, lin1_w, lin1_b, lin2_w, lin2_b,
      in_b, bim_ln_w, bim_ln_b, m_conv_w, m_conv_b, m_dt_w, m_dt_b,
      m_A_log, m_D, post_ln_w, post_ln_b, final_w, final_b,
      ws + WS_INW_OFF, ws + WS_INPROJ_OFF, ws + WS_XPROJ_OFF, ws + WS_OUTPROJ_OFF,
      (float*)d_out);
}
